// NaiveFourierKANLayer_18700287607283
// MI455X (gfx1250) — matrix operand order repaired and checked by reading, not compiled, not run on hardware
//
#include <hip/hip_runtime.h>

// ---------------------------------------------------------------------------
// Fourier-KAN layer as a bf16 WMMA GEMM on gfx1250 (MI455X).
//   y[n,o] = sum_{i,g} cos(x[n,i]*(g+1))*Fc[0,o,i,g] + sin(...)*Fc[1,o,i,g] + bias[o]
// => GEMM with N=8192, O=512, K=8192 where K index k = (i*8+g)*2 + t.
// A generated on the fly (one sincos + Chebyshev-style recurrence per column),
// B packed once to ws. Double-buffered LDS (1 barrier/iter), double-buffered
// B fragments in registers, per-wave tile 32x128 (2 WMMAs per B fragment).
// ---------------------------------------------------------------------------

#define N_TOTAL 8192
#define IDIM    512
#define ODIM    512
#define GRID    8
#define KTOT    (IDIM * GRID * 2)   // 8192
#define NKB     (KTOT / 32)         // 256 K-steps
#define BM      256
#define BO      128
#define BK      32
#define BKP     40                   // padded LDS row stride (ushorts) to spread banks

typedef __attribute__((ext_vector_type(16))) __bf16          v16bf;
typedef __attribute__((ext_vector_type(8)))  float           v8f;
typedef __attribute__((ext_vector_type(8)))  unsigned short  ushort8;

union ABf { v16bf v; ushort8 h[2]; };

__device__ __forceinline__ unsigned short f2bf(float f) {
    union { float f; unsigned u; } v; v.f = f;
    unsigned r = v.u + 0x7FFFu + ((v.u >> 16) & 1u);   // round-to-nearest-even
    return (unsigned short)(r >> 16);
}

// -------------------- pre-pass 1: transpose x -> xT[512][8192] -------------
__global__ __launch_bounds__(256) void transpose_x(const float* __restrict__ x,
                                                   float* __restrict__ xT) {
    __shared__ float tile[32][33];
    const int bi = blockIdx.x;            // along I (512/32 = 16)
    const int bn = blockIdx.y;            // along N (8192/32 = 256)
    const int tx = threadIdx.x & 31;
    const int ty = threadIdx.x >> 5;      // 0..7
    #pragma unroll
    for (int j = 0; j < 32; j += 8)
        tile[ty + j][tx] = x[(size_t)(bn * 32 + ty + j) * IDIM + bi * 32 + tx];
    __syncthreads();
    #pragma unroll
    for (int j = 0; j < 32; j += 8)
        xT[(size_t)(bi * 32 + ty + j) * N_TOTAL + bn * 32 + tx] = tile[tx][ty + j];
}

// -------------------- pre-pass 2: pack coeffs -> Bpk[o][k] bf16 ------------
// Fc flat index: ((t*ODIM + o)*IDIM + i)*GRID + g ; Bpk[o][ (i*8+g)*2 + t ]
__global__ __launch_bounds__(256) void pack_B(const float* __restrict__ fc,
                                              unsigned short* __restrict__ Bpk) {
    const int idx = blockIdx.x * 256 + threadIdx.x;   // o*IDIM + i, 512*512 total
    const int o = idx >> 9;
    const int i = idx & 511;
    const float* c = fc + ((size_t)o * IDIM + i) * GRID;                       // t = 0
    const float* s = fc + (size_t)ODIM * IDIM * GRID + ((size_t)o * IDIM + i) * GRID; // t = 1
    unsigned short tmp[16];
    #pragma unroll
    for (int g = 0; g < GRID; ++g) {
        tmp[2 * g]     = f2bf(c[g]);
        tmp[2 * g + 1] = f2bf(s[g]);
    }
    uint4* dst = (uint4*)(Bpk + (size_t)o * KTOT + i * 16);
    dst[0] = *(uint4*)&tmp[0];
    dst[1] = *(uint4*)&tmp[8];
}

// -------------------- main GEMM ---------------------------------------------
__global__ __launch_bounds__(256) void fkan_wmma_gemm(
        const float* __restrict__ xT,            // [IDIM][N_TOTAL]
        const unsigned short* __restrict__ Bpk,  // [ODIM][KTOT] bf16
        const float* __restrict__ bias,          // [ODIM]
        float* __restrict__ out)                 // [N_TOTAL][ODIM]
{
    __shared__ __align__(16) unsigned short Alds[2][BM * BKP]; // 2 x 20 KB
    __shared__ __align__(16) unsigned short Blds[2][BO * BKP]; // 2 x 10 KB

    const int n0   = blockIdx.x * BM;
    const int o0   = blockIdx.y * BO;
    const int tid  = threadIdx.x;
    const int lane = tid & 31;
    const int wave = tid >> 5;               // 0..7 -> 32-row strip

    v8f acc[2][8];
    #pragma unroll
    for (int f = 0; f < 2; ++f)
        #pragma unroll
        for (int ot = 0; ot < 8; ++ot) acc[f][ot] = (v8f)(0.0f);

    // per-lane fragment addressing (ISA 16-bit A/B VGPR layouts)
    const int frow    = lane & 15;
    const int half    = lane >> 4;           // 0: lanes 0-15, 1: lanes 16-31
    const int a_base0 = (wave * 32 + frow) * BKP + half * 8;        // A frag 0
    const int a_base1 = (wave * 32 + 16 + frow) * BKP + half * 8;   // A frag 1
    const int bkoff   = half * 8;                                    // B: K 8h..8h+7, 16+8h..16+8h+7

    // stage K-step kb into LDS buffer `buf`
    auto stage = [&](int kb, int buf) {
        // ---- A: cos/sin(x*(g+1)) via one sincos + angle-addition recurrence
        #pragma unroll
        for (int q = 0; q < 2; ++q) {
            const int r  = tid;              // row within tile (0..255)
            const int i  = kb * 2 + q;       // input column
            const float xv = xT[(size_t)i * N_TOTAL + n0 + r];
            float s1, c1;
            __sincosf(xv, &s1, &c1);         // v_sin_f32 + v_cos_f32
            unsigned short tmp[16];
            float cg = c1, sg = s1;
            tmp[0] = f2bf(cg);
            tmp[1] = f2bf(sg);
            #pragma unroll
            for (int g = 1; g < GRID; ++g) { // cos/sin((g+1)x) by rotation
                const float cn = __builtin_fmaf(cg, c1, -sg * s1);
                const float sn = __builtin_fmaf(sg, c1,  cg * s1);
                cg = cn; sg = sn;
                tmp[2 * g]     = f2bf(cg);
                tmp[2 * g + 1] = f2bf(sg);
            }
            ushort8* dst = (ushort8*)&Alds[buf][r * BKP + q * 16];
            dst[0] = *(ushort8*)&tmp[0];
            dst[1] = *(ushort8*)&tmp[8];
        }
        // ---- B: packed bf16 coeffs, coalesced uint4 copies
        {
            const int o    = tid >> 1;       // 0..127
            const int part = tid & 1;
            const uint4* src = (const uint4*)(Bpk + (size_t)(o0 + o) * KTOT + kb * BK + part * 16);
            uint4* dst = (uint4*)&Blds[buf][o * BKP + part * 16];
            dst[0] = src[0];
            dst[1] = src[1];
            if (kb + 1 < NKB)                // gfx1250 global_prefetch path
                __builtin_prefetch(Bpk + (size_t)(o0 + o) * KTOT + (kb + 1) * BK + part * 16, 0, 3);
        }
    };

    stage(0, 0);                             // prologue

    for (int kb = 0; kb < NKB; ++kb) {
        const int buf = kb & 1;
        __syncthreads();                     // buf ready; buf^1 free (last read in previous iter)
        if (kb + 1 < NKB) stage(kb + 1, buf ^ 1);   // overlaps with WMMA below

        // ---- compute: 16 WMMAs, B fragments double-buffered in registers ----
        ABf a0, a1;
        a0.h[0] = *(const ushort8*)&Alds[buf][a_base0];
        a0.h[1] = *(const ushort8*)&Alds[buf][a_base0 + 16];
        a1.h[0] = *(const ushort8*)&Alds[buf][a_base1];
        a1.h[1] = *(const ushort8*)&Alds[buf][a_base1 + 16];

        ABf b[2];
        b[0].h[0] = *(const ushort8*)&Blds[buf][frow * BKP + bkoff];
        b[0].h[1] = *(const ushort8*)&Blds[buf][frow * BKP + bkoff + 16];
        #pragma unroll
        for (int ot = 0; ot < 8; ++ot) {
            if (ot + 1 < 8) {                // prefetch next B fragment into regs
                const int nb = ((ot + 1) * 16 + frow) * BKP + bkoff;
                b[(ot + 1) & 1].h[0] = *(const ushort8*)&Blds[buf][nb];
                b[(ot + 1) & 1].h[1] = *(const ushort8*)&Blds[buf][nb + 16];
            }
            acc[0][ot] = __builtin_amdgcn_wmma_f32_16x16x32_bf16(
                false, a0.v, false, b[ot & 1].v, (short)0, acc[0][ot], false, false);
            acc[1][ot] = __builtin_amdgcn_wmma_f32_16x16x32_bf16(
                false, a1.v, false, b[ot & 1].v, (short)0, acc[1][ot], false, false);
        }
    }

    // ---- epilogue: C layout VGPR j -> M = j + 8*half, N = lane%16 ----
    const int mrow = half * 8;
    const int col  = lane & 15;
    #pragma unroll
    for (int ot = 0; ot < 8; ++ot) {
        const int o = o0 + ot * 16 + col;
        const float bv = bias[o];
        #pragma unroll
        for (int f = 0; f < 2; ++f) {
            #pragma unroll
            for (int j = 0; j < 8; ++j) {
                const int n = n0 + wave * 32 + f * 16 + mrow + j;
                out[(size_t)n * ODIM + o] = acc[f][ot][j] + bv;
            }
        }
    }
}

// ---------------------------------------------------------------------------
extern "C" void kernel_launch(void* const* d_in, const int* in_sizes, int n_in,
                              void* d_out, int out_size, void* d_ws, size_t ws_size,
                              hipStream_t stream) {
    const float* x    = (const float*)d_in[0];   // [8192,512]
    const float* fc   = (const float*)d_in[1];   // [2,512,512,8]
    const float* bias = (const float*)d_in[2];   // [1,512]
    float* out        = (float*)d_out;           // [8192,512]

    float* xT            = (float*)d_ws;                                  // 16 MB
    unsigned short* Bpk  = (unsigned short*)((char*)d_ws + (size_t)IDIM * N_TOTAL * 4); // 8 MB

    transpose_x<<<dim3(IDIM / 32, N_TOTAL / 32), 256, 0, stream>>>(x, xT);
    pack_B<<<(ODIM * IDIM) / 256, 256, 0, stream>>>(fc, Bpk);
    fkan_wmma_gemm<<<dim3(N_TOTAL / BM, ODIM / BO), 256, 0, stream>>>(xT, Bpk, bias, out);
}
